// MultiHeadAttention_29343216566518
// MI455X (gfx1250) — matrix operand order repaired and checked by reading, not compiled, not run on hardware
//
#include <hip/hip_runtime.h>
#include <hip/hip_bf16.h>

typedef __bf16 bf16;
typedef __attribute__((ext_vector_type(8)))  bf16  bf16x8;
typedef __attribute__((ext_vector_type(16))) bf16  bf16x16;
typedef __attribute__((ext_vector_type(4)))  float floatx4;
typedef __attribute__((ext_vector_type(8)))  float v8f;

#define B_  2
#define S_  4096
#define D_  512
#define H_  8
#define HD_ 64
#define SCALE_ 0.125f  // HD^-0.5 = 1/8

// LDS layout for the fused attention kernel
#define SROW_      4100                       // 4096 + 4 pad floats (bank-conflict avoidance)
#define SC_BYTES_  (16 * SROW_ * 4)           // 262400: 16 x 4096 fp32 score strip
#define RED_OFF_   SC_BYTES_                  // 1024 fp32 reduction buffer
#define PART_OFF_  (RED_OFF_ + 4096)          // 32 waves x 256 fp32 AV partials
#define SMEM_BYTES_ (PART_OFF_ + 32768)       // 299264 B  (< 320 KB WGP LDS)

__device__ __forceinline__ v8f wmma_bf16(bf16x16 a, bf16x16 b, v8f c) {
  return __builtin_amdgcn_wmma_f32_16x16x32_bf16(false, a, false, b, (short)0, c,
                                                 false, false);
}

__device__ __forceinline__ bf16x16 cat8(bf16x8 lo, bf16x8 hi) {
  bf16x16 r;
#pragma unroll
  for (int j = 0; j < 8; ++j) { r[j] = lo[j]; r[8 + j] = hi[j]; }
  return r;
}

__device__ __forceinline__ bf16x16 cvt16(floatx4 f0, floatx4 f1, floatx4 f2, floatx4 f3) {
  bf16x16 r;
#pragma unroll
  for (int j = 0; j < 4; ++j) {
    r[j]      = (bf16)f0[j];
    r[4 + j]  = (bf16)f1[j];
    r[8 + j]  = (bf16)f2[j];
    r[12 + j] = (bf16)f3[j];
  }
  return r;
}

// ---------------------------------------------------------------------------
// K0: transpose + convert a 512x512 fp32 weight matrix to bf16 (Wt[n][k]=W[k][n])
// ---------------------------------------------------------------------------
__global__ __launch_bounds__(256) void wt_kernel(const float* __restrict__ w,
                                                 bf16* __restrict__ wt) {
  int i = blockIdx.x * 256 + threadIdx.x;  // 512*512 elements
  int n = i >> 9, k = i & 511;
  wt[n * 512 + k] = (bf16)w[k * 512 + n];
}

// ---------------------------------------------------------------------------
// K1: Q/K/V projections. One wave = one 16x16 output tile, K-loop over 512.
//   mat 0 -> Qb [B,H,S,HD] bf16 ; mat 1 -> Kb [B,H,S,HD] ; mat 2 -> Vt [B,H,HD,S]
// ---------------------------------------------------------------------------
__global__ __launch_bounds__(256) void proj_kernel(
    const float* __restrict__ xq, const float* __restrict__ xk,
    const float* __restrict__ xv, const bf16* __restrict__ wtq,
    const bf16* __restrict__ wtk, const bf16* __restrict__ wtv,
    const float* __restrict__ bq, const float* __restrict__ bk,
    const float* __restrict__ bv, bf16* __restrict__ Qb, bf16* __restrict__ Kb,
    bf16* __restrict__ Vt) {
  int wave = blockIdx.x * 8 + (threadIdx.x >> 5);
  int lane = threadIdx.x & 31;
  int mat = wave / 16384;  // 0=q 1=k 2=v
  int t   = wave % 16384;
  int nt  = t & 31;        // 32 n-tiles (D/16)
  int mt  = t >> 5;        // 512 m-tiles (B*S/16)

  const float* X    = mat == 0 ? xq  : (mat == 1 ? xk  : xv);
  const bf16*  Wt   = mat == 0 ? wtq : (mat == 1 ? wtk : wtv);
  const float* bias = mat == 0 ? bq  : (mat == 1 ? bk  : bv);

  int g = lane >> 4;
  int m = lane & 15;
  const float* xrow = X + (long)(mt * 16 + m) * D_;
  const bf16*  wrow = Wt + (long)(nt * 16 + m) * D_;

  v8f c = {};
#pragma unroll 4
  for (int dB = 0; dB < D_; dB += 32) {
    floatx4 f0 = *(const floatx4*)(xrow + dB + 8 * g);
    floatx4 f1 = *(const floatx4*)(xrow + dB + 8 * g + 4);
    floatx4 f2 = *(const floatx4*)(xrow + dB + 16 + 8 * g);
    floatx4 f3 = *(const floatx4*)(xrow + dB + 16 + 8 * g + 4);
    bf16x8 b0 = *(const bf16x8*)(wrow + dB + 8 * g);
    bf16x8 b1 = *(const bf16x8*)(wrow + dB + 16 + 8 * g);
    c = wmma_bf16(cvt16(f0, f1, f2, f3), cat8(b0, b1), c);
  }

  int col = nt * 16 + m;
  float bia = bias[col];
  int h = col >> 6, hd = col & 63;
#pragma unroll
  for (int r = 0; r < 8; ++r) {
    int grow = mt * 16 + r + 8 * g;  // global row in [0, B*S)
    int bb = grow >> 12;
    int s  = grow & 4095;
    bf16 val = (bf16)(c[r] + bia);
    if (mat == 2)
      Vt[((long)((bb * H_ + h) * HD_ + hd)) * S_ + s] = val;
    else if (mat == 0)
      Qb[((long)((bb * H_ + h) * S_ + s)) * HD_ + hd] = val;
    else
      Kb[((long)((bb * H_ + h) * S_ + s)) * HD_ + hd] = val;
  }
}

// ---------------------------------------------------------------------------
// K2 (fused): scores -> softmax -> AV for one 16-query strip of one (b,h).
//   1024 threads = 32 waves; 16 x 4096 fp32 score strip lives in LDS.
//   Normalized weights streamed to d_out with nontemporal stores.
// ---------------------------------------------------------------------------
__global__ __launch_bounds__(1024) void attn_kernel(
    const bf16* __restrict__ Qb, const bf16* __restrict__ Kb,
    const bf16* __restrict__ Vt, float* __restrict__ Wout,
    bf16* __restrict__ Ctx) {
  extern __shared__ char smem[];
  float* sc   = (float*)smem;                 // [16][SROW_]
  float* red  = (float*)(smem + RED_OFF_);    // [1024]
  float* part = (float*)(smem + PART_OFF_);   // [32][256]

  const int t    = threadIdx.x;
  const int wave = t >> 5;
  const int lane = t & 31;
  const int g = lane >> 4, m = lane & 15;

  const int qt = blockIdx.x & 255;
  const int h  = (blockIdx.x >> 8) & 7;
  const int bb = blockIdx.x >> 11;
  const int bh = bb * H_ + h;

  // ---- Phase 1: score tiles (WMMA) into LDS -------------------------------
  bf16x16 aq0, aq1;  // A fragment (Q rows) reused across all key tiles
  {
    const bf16* qrow = Qb + ((long)(bh * S_ + qt * 16 + m)) * HD_;
    bf16x8 a0 = *(const bf16x8*)(qrow + 0 + 8 * g);
    bf16x8 a1 = *(const bf16x8*)(qrow + 16 + 8 * g);
    bf16x8 a2 = *(const bf16x8*)(qrow + 32 + 8 * g);
    bf16x8 a3 = *(const bf16x8*)(qrow + 48 + 8 * g);
    aq0 = cat8(a0, a1);
    aq1 = cat8(a2, a3);
  }
  for (int kt8 = 0; kt8 < 8; ++kt8) {
    int kt = kt8 * 32 + wave;  // 256 key tiles split across 32 waves
    const bf16* krow = Kb + ((long)(bh * S_ + kt * 16 + m)) * HD_;
    bf16x8 b0 = *(const bf16x8*)(krow + 0 + 8 * g);
    bf16x8 b1 = *(const bf16x8*)(krow + 16 + 8 * g);
    bf16x8 b2 = *(const bf16x8*)(krow + 32 + 8 * g);
    bf16x8 b3 = *(const bf16x8*)(krow + 48 + 8 * g);
    v8f c = {};
    c = wmma_bf16(aq0, cat8(b0, b1), c);
    c = wmma_bf16(aq1, cat8(b2, b3), c);
#pragma unroll
    for (int r = 0; r < 8; ++r)
      sc[(r + 8 * g) * SROW_ + kt * 16 + m] = c[r] * SCALE_;
  }
  __syncthreads();

  // ---- Phase 2: softmax over each of the 16 rows (64 threads per row) -----
  const int row = t >> 6;  // 0..15
  const int c64 = t & 63;  // position within row team
  float* prow = sc + row * SROW_;

  float mx = -3.0e38f;
#pragma unroll 4
  for (int i = 0; i < 16; ++i) {
    floatx4 x = *(const floatx4*)(prow + c64 * 4 + i * 256);
    mx = fmaxf(mx, fmaxf(fmaxf(x[0], x[1]), fmaxf(x[2], x[3])));
  }
  red[t] = mx;
  __syncthreads();
  for (int s = 32; s > 0; s >>= 1) {
    if (c64 < s) red[t] = fmaxf(red[t], red[t + s]);
    __syncthreads();
  }
  mx = red[row << 6];
  __syncthreads();

  float sum = 0.f;
#pragma unroll 4
  for (int i = 0; i < 16; ++i) {
    floatx4 x = *(const floatx4*)(prow + c64 * 4 + i * 256);
#pragma unroll
    for (int j = 0; j < 4; ++j) {
      x[j] = __expf(x[j] - mx);
      sum += x[j];
    }
    *(floatx4*)(prow + c64 * 4 + i * 256) = x;
  }
  red[t] = sum;
  __syncthreads();
  for (int s = 32; s > 0; s >>= 1) {
    if (c64 < s) red[t] += red[t + s];
    __syncthreads();
  }
  const float inv = 1.0f / red[row << 6];

  // normalize in LDS + stream final weights to HBM (write-once -> NT hint)
  float* wout = Wout + ((long)(bh * S_ + qt * 16 + row)) * S_;
#pragma unroll 4
  for (int i = 0; i < 16; ++i) {
    floatx4 x = *(const floatx4*)(prow + c64 * 4 + i * 256);
#pragma unroll
    for (int j = 0; j < 4; ++j) x[j] *= inv;
    *(floatx4*)(prow + c64 * 4 + i * 256) = x;
    __builtin_nontemporal_store(x, (floatx4*)(wout + c64 * 4 + i * 256));
  }
  __syncthreads();

  // ---- Phase 3: ctx = weights @ V (A from LDS, 8-way K split per dt tile) --
  const int dt = wave & 3;   // which 16-col slice of HD
  const int ks = wave >> 2;  // which 512-wide K slice
  const bf16* vrow = Vt + ((long)(bh * HD_ + dt * 16 + m)) * S_;
  const int k0 = ks * 512;
  v8f c = {};
#pragma unroll 2
  for (int kB = k0; kB < k0 + 512; kB += 32) {
    floatx4 f0 = *(const floatx4*)(sc + m * SROW_ + kB + 8 * g);
    floatx4 f1 = *(const floatx4*)(sc + m * SROW_ + kB + 8 * g + 4);
    floatx4 f2 = *(const floatx4*)(sc + m * SROW_ + kB + 16 + 8 * g);
    floatx4 f3 = *(const floatx4*)(sc + m * SROW_ + kB + 16 + 8 * g + 4);
    bf16x8 b0 = *(const bf16x8*)(vrow + kB + 8 * g);
    bf16x8 b1 = *(const bf16x8*)(vrow + kB + 16 + 8 * g);
    c = wmma_bf16(cvt16(f0, f1, f2, f3), cat8(b0, b1), c);
  }
#pragma unroll
  for (int r = 0; r < 8; ++r) part[wave * 256 + r * 32 + lane] = c[r];
  __syncthreads();

  if (wave < 4) {  // wave == dt: reduce the 8 K-slices and store ctx
    v8f acc = {};
#pragma unroll
    for (int k8 = 0; k8 < 8; ++k8)
#pragma unroll
      for (int r = 0; r < 8; ++r)
        acc[r] += part[(wave + 4 * k8) * 256 + r * 32 + lane];
    int d = h * 64 + wave * 16 + m;
#pragma unroll
    for (int r = 0; r < 8; ++r) {
      int s = qt * 16 + r + 8 * g;
      Ctx[((long)(bb * S_ + s)) * D_ + d] = (bf16)acc[r];
    }
  }
}

// ---------------------------------------------------------------------------
// K3: out = ctx @ wo + bo  (fp32 result into d_out).
// ---------------------------------------------------------------------------
__global__ __launch_bounds__(256) void outproj_kernel(const bf16* __restrict__ Ctx,
                                                      const bf16* __restrict__ WtO,
                                                      const float* __restrict__ bo,
                                                      float* __restrict__ Out) {
  int wave = blockIdx.x * 8 + (threadIdx.x >> 5);
  int lane = threadIdx.x & 31;
  int nt = wave & 31;
  int mt = wave >> 5;
  int g = lane >> 4, m = lane & 15;

  const bf16* arow = Ctx + (long)(mt * 16 + m) * D_;
  const bf16* brow = WtO + (long)(nt * 16 + m) * D_;

  v8f c = {};
#pragma unroll 4
  for (int dB = 0; dB < D_; dB += 32) {
    bf16x8 a0 = *(const bf16x8*)(arow + dB + 8 * g);
    bf16x8 a1 = *(const bf16x8*)(arow + dB + 16 + 8 * g);
    bf16x8 b0 = *(const bf16x8*)(brow + dB + 8 * g);
    bf16x8 b1 = *(const bf16x8*)(brow + dB + 16 + 8 * g);
    c = wmma_bf16(cat8(a0, a1), cat8(b0, b1), c);
  }

  float bia = bo[nt * 16 + m];
#pragma unroll
  for (int r = 0; r < 8; ++r)
    Out[(long)(mt * 16 + r + 8 * g) * D_ + nt * 16 + m] = c[r] + bia;
}

// ---------------------------------------------------------------------------
extern "C" void kernel_launch(void* const* d_in, const int* in_sizes, int n_in,
                              void* d_out, int out_size, void* d_ws, size_t ws_size,
                              hipStream_t stream) {
  const float* query = (const float*)d_in[0];
  const float* key   = (const float*)d_in[1];
  const float* value = (const float*)d_in[2];
  const float* wq    = (const float*)d_in[3];
  const float* bq    = (const float*)d_in[4];
  const float* wk    = (const float*)d_in[5];
  const float* bk    = (const float*)d_in[6];
  const float* wv    = (const float*)d_in[7];
  const float* bv    = (const float*)d_in[8];
  const float* wo    = (const float*)d_in[9];
  const float* bo    = (const float*)d_in[10];

  float* out     = (float*)d_out;               // [B,S,D]
  float* weights = out + (size_t)B_ * S_ * D_;  // [B,H,S,S]

  char* ws = (char*)d_ws;
  const size_t WT_BYTES  = (size_t)512 * 512 * 2;     // one bf16 512x512
  const size_t BSD_BYTES = (size_t)B_ * S_ * D_ * 2;  // bf16 [B,S,D]
  bf16* wtq = (bf16*)(ws + 0 * WT_BYTES);
  bf16* wtk = (bf16*)(ws + 1 * WT_BYTES);
  bf16* wtv = (bf16*)(ws + 2 * WT_BYTES);
  bf16* wto = (bf16*)(ws + 3 * WT_BYTES);
  char* big = ws + 4 * WT_BYTES;
  bf16* Qb  = (bf16*)(big + 0 * BSD_BYTES);
  bf16* Kb  = (bf16*)(big + 1 * BSD_BYTES);
  bf16* Vt  = (bf16*)(big + 2 * BSD_BYTES);  // [B,H,HD,S]
  bf16* Ctx = (bf16*)(big + 3 * BSD_BYTES);  // [B,S,D]

  wt_kernel<<<1024, 256, 0, stream>>>(wq, wtq);
  wt_kernel<<<1024, 256, 0, stream>>>(wk, wtk);
  wt_kernel<<<1024, 256, 0, stream>>>(wv, wtv);
  wt_kernel<<<1024, 256, 0, stream>>>(wo, wto);

  proj_kernel<<<6144, 256, 0, stream>>>(query, key, value, wtq, wtk, wtv, bq, bk,
                                        bv, Qb, Kb, Vt);
  attn_kernel<<<4096, 1024, SMEM_BYTES_, stream>>>(Qb, Kb, Vt, weights, Ctx);
  outproj_kernel<<<2048, 256, 0, stream>>>(Ctx, wto, bo, out);
}
